// Qwen3MoeAttention_89953795048058
// MI455X (gfx1250) — compile-verified, matrix operand order checked by reading, not run on hardware
//
#include <hip/hip_runtime.h>

// ---------------- problem constants ----------------
constexpr int B_  = 4;
constexpr int S_  = 1024;
constexpr int H_  = 2048;
constexpr int NQ_ = 32;
constexpr int NKV_= 4;
constexpr int D_  = 128;
constexpr int E_  = (NQ_ + 2*NKV_) * D_;   // 5120
constexpr int M_  = B_ * S_;               // 4096
constexpr float EPS_   = 1e-6f;
constexpr float SCALE_ = 0.08838834764831845f; // 1/sqrt(128)

// ---------------- WMMA types ----------------
typedef __bf16  v16bf __attribute__((ext_vector_type(16)));
typedef float   v8f   __attribute__((ext_vector_type(8)));

union AFrag {
    v16bf v;
    uint4 q[2];
    unsigned short u[16];
};

__device__ inline v8f wmma_bf16(v16bf a, v16bf b, v8f c) {
    return __builtin_amdgcn_wmma_f32_16x16x32_bf16(false, a, false, b, (short)0, c, false, false);
}

__device__ inline unsigned short f32_to_bf16(float f) {
    unsigned int u = __float_as_uint(f);
    unsigned int r = u + 0x7fffu + ((u >> 16) & 1u);
    return (unsigned short)(r >> 16);
}

// ---------------- cast f32 -> bf16 (vector4) ----------------
__global__ __launch_bounds__(256) void cast_f32_bf16(const float* __restrict__ in,
                                                     unsigned short* __restrict__ out,
                                                     size_t n4) {
    size_t i = (size_t)blockIdx.x * blockDim.x + threadIdx.x;
    if (i >= n4) return;
    float4 v = reinterpret_cast<const float4*>(in)[i];
    unsigned short o[4] = { f32_to_bf16(v.x), f32_to_bf16(v.y),
                            f32_to_bf16(v.z), f32_to_bf16(v.w) };
    reinterpret_cast<ushort2*>(out)[2*i + 0] = make_ushort2(o[0], o[1]);
    reinterpret_cast<ushort2*>(out)[2*i + 1] = make_ushort2(o[2], o[3]);
}

// ---------------- transpose + cast: in (R x C) f32 -> out (C x R) bf16 ----------------
__global__ __launch_bounds__(256) void transpose_cast_bf16(const float* __restrict__ in,
                                                           unsigned short* __restrict__ out,
                                                           int R, int C) {
    __shared__ float tile[32][33];
    int c0 = blockIdx.x * 32;
    int r0 = blockIdx.y * 32;
    int tx = threadIdx.x, ty = threadIdx.y; // block (32,8)
#pragma unroll
    for (int i = 0; i < 4; ++i) {
        int r = r0 + ty + i*8, c = c0 + tx;
        tile[ty + i*8][tx] = (r < R && c < C) ? in[(size_t)r * C + c] : 0.f;
    }
    __syncthreads();
#pragma unroll
    for (int i = 0; i < 4; ++i) {
        int c = c0 + ty + i*8, r = r0 + tx;
        if (c < C && r < R)
            out[(size_t)c * R + r] = f32_to_bf16(tile[tx][ty + i*8]);
    }
}

// ---------------- bf16 GEMM: C(MxN,f32) = A(MxK,bf16 rowmaj) * Bt(NxK,bf16 rowmaj)^T ----------------
// 4 waves / block covering a 64x128 C tile; each wave owns 32x64 (2x4 WMMA fragments).
__global__ __launch_bounds__(128) void gemm_bf16_nt(const unsigned short* __restrict__ A,
                                                    const unsigned short* __restrict__ Bt,
                                                    float* __restrict__ C,
                                                    int M, int N, int K) {
    int wave = threadIdx.x >> 5;
    int lane = threadIdx.x & 31;
    int half = lane >> 4;
    int lid  = lane & 15;
    int nbn = N >> 7;
    int bm = blockIdx.x / nbn;
    int bn = blockIdx.x % nbn;
    int m0 = bm * 64  + (wave >> 1) * 32;
    int n0 = bn * 128 + (wave & 1) * 64;

    const v8f z = {0.f,0.f,0.f,0.f,0.f,0.f,0.f,0.f};
    v8f acc[2][4];
#pragma unroll
    for (int i = 0; i < 2; ++i)
#pragma unroll
        for (int j = 0; j < 4; ++j) acc[i][j] = z;

    const unsigned short* aRow[2] = { A  + (size_t)(m0 +      lid) * K,
                                      A  + (size_t)(m0 + 16 + lid) * K };
    const unsigned short* bRow[4] = { Bt + (size_t)(n0 +      lid) * K,
                                      Bt + (size_t)(n0 + 16 + lid) * K,
                                      Bt + (size_t)(n0 + 32 + lid) * K,
                                      Bt + (size_t)(n0 + 48 + lid) * K };

    for (int k0 = 0; k0 < K; k0 += 32) {
        int off = k0 + half * 8;
        // L2 prefetch of the stream ~512B ahead (global_prefetch_b8)
        __builtin_prefetch(aRow[0] + off + 256, 0, 1);
        __builtin_prefetch(aRow[1] + off + 256, 0, 1);
        __builtin_prefetch(bRow[0] + off + 256, 0, 1);
        __builtin_prefetch(bRow[1] + off + 256, 0, 1);
        __builtin_prefetch(bRow[2] + off + 256, 0, 1);
        __builtin_prefetch(bRow[3] + off + 256, 0, 1);

        AFrag a[2], b[4];
#pragma unroll
        for (int t = 0; t < 2; ++t) {
            a[t].q[0] = *reinterpret_cast<const uint4*>(aRow[t] + off);
            a[t].q[1] = *reinterpret_cast<const uint4*>(aRow[t] + off + 16);
        }
#pragma unroll
        for (int t = 0; t < 4; ++t) {
            b[t].q[0] = *reinterpret_cast<const uint4*>(bRow[t] + off);
            b[t].q[1] = *reinterpret_cast<const uint4*>(bRow[t] + off + 16);
        }
#pragma unroll
        for (int i = 0; i < 2; ++i)
#pragma unroll
            for (int j = 0; j < 4; ++j)
                acc[i][j] = wmma_bf16(a[i].v, b[j].v, acc[i][j]);
    }

#pragma unroll
    for (int ti = 0; ti < 2; ++ti)
#pragma unroll
        for (int tj = 0; tj < 4; ++tj)
#pragma unroll
            for (int r = 0; r < 8; ++r) {
                int row = m0 + ti*16 + r + 8*half;
                int col = n0 + tj*16 + lid;
                C[(size_t)row * N + col] = acc[ti][tj][r];
            }
}

// ---------------- RMSNorm + RoPE, split Q/K, V transposed, cast bf16 ----------------
// One wave per (b,s,head); head 0..39 = [q:32][k:4][v:4]. Lane owns d = lane*4..+3.
// V is written TRANSPOSED: Vt[b][kv][d][s] so attention P*V fragments load contiguously.
__global__ __launch_bounds__(128) void qkv_norm_rope(const float* __restrict__ qkv,
                                                     const float* __restrict__ qw,
                                                     const float* __restrict__ kw,
                                                     const float* __restrict__ cosb,
                                                     const float* __restrict__ sinb,
                                                     unsigned short* __restrict__ Q,
                                                     unsigned short* __restrict__ Kd,
                                                     unsigned short* __restrict__ Vt) {
    int wave = threadIdx.x >> 5;
    int lane = threadIdx.x & 31;
    int inst = blockIdx.x * 4 + wave;           // < M_*40
    int row  = inst / 40;
    int hh   = inst % 40;
    int b = row >> 10;                          // /S_
    int s = row & (S_ - 1);

    float4 xv = *reinterpret_cast<const float4*>(qkv + (size_t)row * E_ + hh * D_ + lane * 4);

    if (hh < NQ_ + NKV_) {
        float ss = xv.x*xv.x + xv.y*xv.y + xv.z*xv.z + xv.w*xv.w;
#pragma unroll
        for (int m = 16; m >= 1; m >>= 1) ss += __shfl_xor(ss, m, 32);
        float inv = rsqrtf(ss * (1.0f / D_) + EPS_);
        const float* w = (hh < NQ_) ? qw : kw;
        float4 wv = *reinterpret_cast<const float4*>(w + lane * 4);
        float xn[4] = { xv.x*inv*wv.x, xv.y*inv*wv.y, xv.z*inv*wv.z, xv.w*inv*wv.w };

        float4 cv = *reinterpret_cast<const float4*>(cosb + (size_t)row * D_ + lane * 4);
        float4 sv = *reinterpret_cast<const float4*>(sinb + (size_t)row * D_ + lane * 4);
        float sgn = (lane < 16) ? -1.f : 1.f;   // d<64 takes -x[d+64], d>=64 takes +x[d-64]
        float rn[4], o[4];
#pragma unroll
        for (int j = 0; j < 4; ++j) rn[j] = __shfl_xor(xn[j], 16, 32);
        float cc[4] = { cv.x, cv.y, cv.z, cv.w };
        float sc[4] = { sv.x, sv.y, sv.z, sv.w };
#pragma unroll
        for (int j = 0; j < 4; ++j) o[j] = xn[j]*cc[j] + sgn*rn[j]*sc[j];

        unsigned short* dst;
        if (hh < NQ_) dst = Q  + (((size_t)b*NQ_  + hh)        * S_ + s) * D_ + lane*4;
        else          dst = Kd + (((size_t)b*NKV_ + (hh-NQ_))  * S_ + s) * D_ + lane*4;
#pragma unroll
        for (int j = 0; j < 4; ++j) dst[j] = f32_to_bf16(o[j]);
    } else {
        int hv = hh - (NQ_ + NKV_);
        float o[4] = { xv.x, xv.y, xv.z, xv.w };
        unsigned short* dstT = Vt + (((size_t)b*NKV_ + hv) * D_ + lane*4) * S_ + s;
#pragma unroll
        for (int j = 0; j < 4; ++j) dstT[(size_t)j * S_] = f32_to_bf16(o[j]);
    }
}

// ---------------- Flash attention (causal, GQA) ----------------
// Block = 4 independent waves; each wave owns 16 query rows of one (b,h).
// K fragments and Vt fragments both load contiguously from global (no LDS staging).
// LDS only holds the per-wave P tile (C-layout -> A-layout lane permutation).
__global__ __launch_bounds__(128) void attn_fwd(const unsigned short* __restrict__ Q,
                                                const unsigned short* __restrict__ Kd,
                                                const unsigned short* __restrict__ Vt,
                                                unsigned short* __restrict__ Oa) {
    __shared__ alignas(16) unsigned short pls[4][16][32];  // per-wave P tile [row][key]

    int wave = threadIdx.x >> 5;
    int lane = threadIdx.x & 31;
    int half = lane >> 4;
    int lid  = lane & 15;

    int bh = blockIdx.x / (S_ / 64);
    int qb = (blockIdx.x % (S_ / 64)) * 64;
    int b  = bh / NQ_;
    int h  = bh % NQ_;
    int kh = h / (NQ_ / NKV_);
    int q0 = qb + wave * 16;

    const unsigned short* Qb  = Q  + ((size_t)b*NQ_  + h ) * S_ * D_;  // [s][d]
    const unsigned short* Kb  = Kd + ((size_t)b*NKV_ + kh) * S_ * D_;  // [s][d]
    const unsigned short* Vtb = Vt + ((size_t)b*NKV_ + kh) * D_ * S_;  // [d][s]

    // Q fragments: 16 rows x 128 d, kept in registers, reused for every key block.
    AFrag qf[4];
    const unsigned short* qrow = Qb + (size_t)(q0 + lid) * D_;
#pragma unroll
    for (int d = 0; d < 4; ++d) {
        int off = d*32 + half*8;
        qf[d].q[0] = *reinterpret_cast<const uint4*>(qrow + off);
        qf[d].q[1] = *reinterpret_cast<const uint4*>(qrow + off + 16);
    }

    const v8f z = {0.f,0.f,0.f,0.f,0.f,0.f,0.f,0.f};
    v8f   o[8]  = {z,z,z,z,z,z,z,z};          // 16 rows x 128 d accumulator
    float mrow[8], lrow[8];
#pragma unroll
    for (int r = 0; r < 8; ++r) { mrow[r] = -1e30f; lrow[r] = 0.f; }

    int rmax = q0 + 15;
    int nkb  = rmax / 32 + 1;                 // wave-private causal range

    for (int kb = 0; kb < nkb; ++kb) {
        int kbase = kb * 32;

        // ---- scores: Q(16x128) x K^T(128x32) via 8 WMMA ----
        v8f c0 = z, c1 = z;
        const unsigned short* kr0 = Kb + (size_t)(kbase +      lid) * D_;
        const unsigned short* kr1 = Kb + (size_t)(kbase + 16 + lid) * D_;
        __builtin_prefetch(kr0 + 32 * D_, 0, 1);   // next key block (global_prefetch_b8)
        __builtin_prefetch(kr1 + 32 * D_, 0, 1);
#pragma unroll
        for (int d = 0; d < 4; ++d) {
            int off = d*32 + half*8;
            AFrag k0, k1;
            k0.q[0] = *reinterpret_cast<const uint4*>(kr0 + off);
            k0.q[1] = *reinterpret_cast<const uint4*>(kr0 + off + 16);
            k1.q[0] = *reinterpret_cast<const uint4*>(kr1 + off);
            k1.q[1] = *reinterpret_cast<const uint4*>(kr1 + off + 16);
            c0 = wmma_bf16(qf[d].v, k0.v, c0);
            c1 = wmma_bf16(qf[d].v, k1.v, c1);
        }

        // ---- online softmax; C-layout row = r + 8*half, col = lid ----
        float p0[8], p1[8], scl[8];
#pragma unroll
        for (int r = 0; r < 8; ++r) {
            int row = q0 + r + 8*half;
            float s0 = c0[r] * SCALE_; if (kbase +      lid > row) s0 = -1e30f;
            float s1 = c1[r] * SCALE_; if (kbase + 16 + lid > row) s1 = -1e30f;
            float mx = fmaxf(s0, s1);
#pragma unroll
            for (int m = 8; m >= 1; m >>= 1) mx = fmaxf(mx, __shfl_xor(mx, m, 32));
            float mnew = fmaxf(mrow[r], mx);
            float sc   = __expf(mrow[r] - mnew);
            float e0 = __expf(s0 - mnew);
            float e1 = __expf(s1 - mnew);
            float sum = e0 + e1;
#pragma unroll
            for (int m = 8; m >= 1; m >>= 1) sum += __shfl_xor(sum, m, 32);
            lrow[r] = lrow[r] * sc + sum;
            mrow[r] = mnew;
            scl[r] = sc; p0[r] = e0; p1[r] = e1;
        }

        // P -> LDS (C-layout write), rescale O
#pragma unroll
        for (int r = 0; r < 8; ++r) {
            pls[wave][r + 8*half][lid]      = f32_to_bf16(p0[r]);
            pls[wave][r + 8*half][16 + lid] = f32_to_bf16(p1[r]);
#pragma unroll
            for (int nt = 0; nt < 8; ++nt) o[nt][r] *= scl[r];
        }

        // P A-fragment (row = lid, key pattern per half)
        AFrag pf;
        pf.q[0] = *reinterpret_cast<const uint4*>(&pls[wave][lid][8*half]);
        pf.q[1] = *reinterpret_cast<const uint4*>(&pls[wave][lid][16 + 8*half]);

        // V^T B-fragments straight from global: Vt[d][s], keys contiguous
#pragma unroll
        for (int nt = 0; nt < 8; ++nt) {
            const unsigned short* vrow = Vtb + (size_t)(nt*16 + lid) * S_ + kbase + 8*half;
            AFrag vf;
            vf.q[0] = *reinterpret_cast<const uint4*>(vrow);
            vf.q[1] = *reinterpret_cast<const uint4*>(vrow + 16);
            o[nt] = wmma_bf16(pf.v, vf.v, o[nt]);
        }
    }

    // epilogue: O /= l, store bf16 into attn[b*S+row][h*D + d]
#pragma unroll
    for (int nt = 0; nt < 8; ++nt)
#pragma unroll
        for (int r = 0; r < 8; ++r) {
            int row = q0 + r + 8*half;
            float val = o[nt][r] / lrow[r];
            Oa[(size_t)(b*S_ + row) * (NQ_*D_) + h*D_ + nt*16 + lid] = f32_to_bf16(val);
        }
}

// ---------------- workspace layout (bytes) ----------------
constexpr size_t OFF_HID  = 0;
constexpr size_t SZ_HID   = (size_t)M_ * H_ * 2;
constexpr size_t OFF_WQT  = OFF_HID + SZ_HID;
constexpr size_t SZ_WQT   = (size_t)E_ * H_ * 2;
constexpr size_t OFF_WOT  = OFF_WQT + SZ_WQT;
constexpr size_t SZ_WOT   = (size_t)H_ * NQ_ * D_ * 2;
constexpr size_t OFF_QKV  = OFF_WOT + SZ_WOT;
constexpr size_t SZ_QKV   = (size_t)M_ * E_ * 4;
constexpr size_t OFF_Q    = OFF_QKV + SZ_QKV;
constexpr size_t SZ_Q     = (size_t)B_ * NQ_ * S_ * D_ * 2;
constexpr size_t OFF_K    = OFF_Q + SZ_Q;
constexpr size_t SZ_K     = (size_t)B_ * NKV_ * S_ * D_ * 2;
constexpr size_t OFF_VT   = OFF_K + SZ_K;
constexpr size_t SZ_VT    = (size_t)B_ * NKV_ * D_ * S_ * 2;
constexpr size_t OFF_ATT  = OFF_VT + SZ_VT;

extern "C" void kernel_launch(void* const* d_in, const int* in_sizes, int n_in,
                              void* d_out, int out_size, void* d_ws, size_t ws_size,
                              hipStream_t stream) {
    (void)in_sizes; (void)n_in; (void)out_size; (void)ws_size;
    const float* hidden = (const float*)d_in[0];
    const float* w_qkv  = (const float*)d_in[1];
    const float* w_o    = (const float*)d_in[2];
    const float* qnw    = (const float*)d_in[3];
    const float* knw    = (const float*)d_in[4];
    const float* cosb   = (const float*)d_in[5];
    const float* sinb   = (const float*)d_in[6];
    float* out = (float*)d_out;

    char* ws = (char*)d_ws;
    unsigned short* hid_bf = (unsigned short*)(ws + OFF_HID);
    unsigned short* wqkvT  = (unsigned short*)(ws + OFF_WQT);
    unsigned short* woT    = (unsigned short*)(ws + OFF_WOT);
    float*          qkv    = (float*)         (ws + OFF_QKV);
    unsigned short* Qb     = (unsigned short*)(ws + OFF_Q);
    unsigned short* Kb     = (unsigned short*)(ws + OFF_K);
    unsigned short* Vtb    = (unsigned short*)(ws + OFF_VT);
    unsigned short* attn   = (unsigned short*)(ws + OFF_ATT);

    // 1) casts / weight transposes
    {
        size_t n4 = (size_t)M_ * H_ / 4;
        cast_f32_bf16<<<(unsigned)((n4 + 255) / 256), 256, 0, stream>>>(hidden, hid_bf, n4);
    }
    {
        dim3 tb(32, 8);
        transpose_cast_bf16<<<dim3(E_/32, H_/32), tb, 0, stream>>>(w_qkv, wqkvT, H_, E_);
        transpose_cast_bf16<<<dim3(H_/32, (NQ_*D_)/32), tb, 0, stream>>>(w_o, woT, NQ_*D_, H_);
    }
    // 2) QKV projection: qkv(M x E) = hidden(M x H) * w_qkv(H x E)
    gemm_bf16_nt<<<(M_/64) * (E_/128), 128, 0, stream>>>(hid_bf, wqkvT, qkv, M_, E_, H_);
    // 3) RMSNorm + RoPE + head split (V transposed)
    qkv_norm_rope<<<(M_ * 40) / 4, 128, 0, stream>>>(qkv, qnw, knw, cosb, sinb, Qb, Kb, Vtb);
    // 4) causal GQA flash attention
    attn_fwd<<<B_ * NQ_ * (S_/64), 128, 0, stream>>>(Qb, Kb, Vtb, attn);
    // 5) output projection: out(M x H) = attn(M x 4096) * w_o(4096 x H)
    gemm_bf16_nt<<<(M_/64) * (H_/128), 128, 0, stream>>>(attn, woT, out, M_, H_, NQ_*D_);
}
